// ERNNCell_21457656611066
// MI455X (gfx1250) — matrix operand order repaired and checked by reading, not compiled, not run on hardware
//
#include <hip/hip_runtime.h>

// ERNN scan for MI455X (gfx1250, wave32, WMMA).
//
// ~120 GFLOP over 512 serial steps. Batch rows are independent: 32 persistent
// workgroups (16 rows = one WMMA M tile each) run the whole scan with only
// workgroup barriers between the 3 GEMM phases per step (3 barriers/step; the
// x_{t+1} stage is folded into phase C so its global-load latency overlaps
// WMMA work).
//
// Bandwidth: each WG needs its 448 KB bf16 weight slice every step; streaming
// it from L2 (~1.5us/step) would be ~5x the XDL time (~0.35us). So weights are
// kept on-chip: Whid/W2/W3 are VGPR-resident (24 B-fragments = 192 VGPRs per
// wave, 16 waves x 16 output cols each), and Win lives in LDS (gfx1250 LDS is
// 320 KB/WG; we use ~113 KB -> one WG per WGP). The inner loop is pure
// ds_load_b128 + v_wmma.
//
// bf16 operands / f32 accumulation. The A operand buffer a1s holds the bf16
// 2*h shadow (cols 0..255, refreshed in phase C) and the staged x_t tile
// (cols 256..383). W_out is pre-scaled by 0.5 so the epilogue reuses the
// 2h shadow.

typedef __attribute__((ext_vector_type(16))) __bf16 v16bf;
typedef __attribute__((ext_vector_type(8)))  float  v8f;

union Frag {
    v16bf bf;
    uint4 q[2];
};

// ---- workspace layout (bf16 elements) ----
#define WS_WHID 0
#define WS_W2   65536
#define WS_W3   131072
#define WS_WIN  196608
#define WS_WOUT 229376
#define WS_TOTAL 262144   // 512 KiB

__global__ void prep_weights(const float* __restrict__ Whid,
                             const float* __restrict__ W2,
                             const float* __restrict__ W3,
                             const float* __restrict__ Win,
                             const float* __restrict__ Wout,
                             __bf16* __restrict__ wsb) {
    int i = blockIdx.x * 256 + threadIdx.x;
    if (i < 65536)        wsb[i] = (__bf16)Whid[i];
    else if (i < 131072)  wsb[i] = (__bf16)W2[i - 65536];
    else if (i < 196608)  wsb[i] = (__bf16)W3[i - 131072];
    else if (i < 229376)  wsb[i] = (__bf16)Win[i - 196608];
    else if (i < 262144)  wsb[i] = (__bf16)(0.5f * Wout[i - 229376]); // fold 1/2 for 2h shadow
}

// LDS strides (elements): row bases stay 16B aligned, 16-row accesses fan out
// across the 64 LDS banks (stride % 64 dwords == 4).
#define HS_STRIDE  260   // fp32 h             : 1040 B/row
#define A1_STRIDE  392   // bf16 [2h | x] rows : 784 B/row (196 dw % 64 == 4)
#define AS_STRIDE  264   // bf16 f1/f2 tiles   : 528 B/row
#define WIN_STRIDE 136   // bf16 Win rows      : 272 B/row
#define XCOL 256         // x region starts at col 256 of a1s

// A-fragment (16x32 bf16): lane reads row (lane&15), chunks [k0..k0+7] and
// [k0+16..k0+23], k0 = kt*32 + 8*(lane>>4).
__device__ __forceinline__ void loadA(Frag& a, const __bf16* ap) {
    a.q[0] = *(const uint4*)(ap);
    a.q[1] = *(const uint4*)(ap + 16);
}

// B-fragment (32x16 bf16): lane reads weight row j0+(lane&15), chunks
// [k0..k0+7] and [k0+16..k0+23], k0 = kt*32 + 8*(lane>>4).
__device__ __forceinline__ void loadB(Frag& b, const __bf16* wp) {
    const uint4* p = (const uint4*)wp;
    b.q[0] = p[0];
    b.q[1] = p[2];
}

#define WMMA(acc, a, b) \
    acc = __builtin_amdgcn_wmma_f32_16x16x32_bf16(false, (a).bf, false, (b).bf, \
                                                  (short)0, acc, false, false)

__global__ __launch_bounds__(512)
void ernn_scan(const float* __restrict__ x0, const float* __restrict__ h0,
               const float* __restrict__ b_in, const float* __restrict__ b_hid,
               const float* __restrict__ b2, const float* __restrict__ b3,
               const float* __restrict__ b_out,
               const __bf16* __restrict__ wsb,
               float* __restrict__ out) {
    __shared__ float  hs  [16 * HS_STRIDE];    // fp32 master copy of h
    __shared__ __bf16 a1s [16 * A1_STRIDE];    // bf16 [2h shadow | x_t] (A operand)
    __shared__ __bf16 f1s [16 * AS_STRIDE];
    __shared__ __bf16 f2s [16 * AS_STRIDE];
    __shared__ __bf16 wins[256 * WIN_STRIDE];  // Win staged in LDS (~68 KB)

    const int tid  = threadIdx.x;
    const int wave = tid >> 5;        // 16 waves; wave w owns output cols [16w, 16w+16)
    const int lane = tid & 31;
    const int half = lane >> 4;
    const int l15  = lane & 15;
    const int b0   = blockIdx.x * 16; // 16 batch rows per workgroup

    const int j0 = wave * 16;
    const int kA = half * 8;    // A-chunk base within a 32-wide k tile
    const int kB = half * 8;    // B-chunk base within a 32-wide k tile

    // ---- VGPR-resident weight slices for the three K=256 GEMMs ----
    Frag wHid[8], w2f[8], w3f[8];
    {
        const __bf16* p = wsb + WS_WHID + (size_t)(j0 + l15) * 256 + kB;
        #pragma unroll
        for (int kt = 0; kt < 8; ++kt) loadB(wHid[kt], p + kt * 32);
        p = wsb + WS_W2 + (size_t)(j0 + l15) * 256 + kB;
        #pragma unroll
        for (int kt = 0; kt < 8; ++kt) loadB(w2f[kt], p + kt * 32);
        p = wsb + WS_W3 + (size_t)(j0 + l15) * 256 + kB;
        #pragma unroll
        for (int kt = 0; kt < 8; ++kt) loadB(w3f[kt], p + kt * 32);
    }

    // ---- stage Win into LDS (row-major, padded stride) ----
    for (int i = tid * 8; i < 256 * 128; i += 512 * 8) {
        int r = i >> 7, c = i & 127;
        *(uint4*)&wins[r * WIN_STRIDE + c] = *(const uint4*)&wsb[WS_WIN + i];
    }

    // loop-invariant biases
    const float bjA = b_in[j0 + l15] + b_hid[j0 + l15];
    const float bj2 = b2[j0 + l15];
    const float bj3 = b3[j0 + l15];

    // loop-invariant pointers / addresses
    const __bf16* aH   = a1s  + l15 * A1_STRIDE + kA;       // A rows (2h | x)
    const __bf16* aF1  = f1s  + l15 * AS_STRIDE + kA;
    const __bf16* aF2  = f2s  + l15 * AS_STRIDE + kA;
    const __bf16* wInL = wins + (j0 + l15) * WIN_STRIDE + kB;

    // precomputed x-staging addresses: per-thread byte offset in x0 (sans t)
    // and destination element index in a1s; the only per-step component is
    // the uniform t*4 scalar base.
    int xi[4], xl[4];
    #pragma unroll
    for (int k = 0; k < 4; ++k) {
        int i = tid + k * 512;            // 2048 elements, 512 threads
        int r = i >> 7, c = i & 127;
        xi[k] = (((b0 + r) * 128 + c) * 512) * 4;   // x0 index: (b*128+f)*512 + t
        xl[k] = r * A1_STRIDE + XCOL + c;
    }

    // load h0 block (fp32 master + bf16 2h shadow) and stage x_0
    for (int i = tid; i < 16 * 256; i += 512) {
        int r = i >> 8, c = i & 255;
        float h = h0[(size_t)(b0 + r) * 256 + c];
        hs [r * HS_STRIDE + c] = h;
        a1s[r * A1_STRIDE + c] = (__bf16)(2.0f * h);
    }
    #pragma unroll
    for (int k = 0; k < 4; ++k)
        a1s[xl[k]] = (__bf16)*(const float*)((const char*)x0 + xi[k]);
    __syncthreads();

    for (int t = 0; t < 512; ++t) {
        // ---- Phase A: f1 = relu(2h @ Whid^T + x @ Win^T + b_hid + b_in) ----
        {
            v8f acc = {};
            #pragma unroll
            for (int kt = 0; kt < 8; ++kt) {   // K = 256 over 2h
                Frag a;
                loadA(a, aH + kt * 32);
                WMMA(acc, a, wHid[kt]);
            }
            #pragma unroll
            for (int kt = 0; kt < 4; ++kt) {   // K = 128 over x, Win from LDS
                Frag a, b;
                loadA(a, aH + XCOL + kt * 32);
                loadB(b, wInL + kt * 32);
                WMMA(acc, a, b);
            }
            #pragma unroll
            for (int d = 0; d < 8; ++d) {
                float v = acc[d] + bjA; v = v > 0.f ? v : 0.f;
                f1s[(d + half * 8) * AS_STRIDE + j0 + l15] = (__bf16)v;
            }
        }
        __syncthreads();

        // ---- Phase B: f2 = relu(f1 @ W2^T + b2) ----
        {
            v8f acc = {};
            #pragma unroll
            for (int kt = 0; kt < 8; ++kt) {
                Frag a;
                loadA(a, aF1 + kt * 32);
                WMMA(acc, a, w2f[kt]);
            }
            #pragma unroll
            for (int d = 0; d < 8; ++d) {
                float v = acc[d] + bj2; v = v > 0.f ? v : 0.f;
                f2s[(d + half * 8) * AS_STRIDE + j0 + l15] = (__bf16)v;
            }
        }
        __syncthreads();

        // ---- Phase C: fn = relu(f2 @ W3^T + b3);
        //      h = 0.98*h + 0.01*fn; refresh 2h shadow; stage x_{t+1} ----
        {
            v8f acc = {};
            #pragma unroll
            for (int kt = 0; kt < 8; ++kt) {
                Frag a;
                loadA(a, aF2 + kt * 32);
                WMMA(acc, a, w3f[kt]);
            }
            // stage x_{t+1} while the WMMA results settle (overlaps latency;
            // x cols of a1s are only read by phase A after the barrier below)
            if (t < 511) {
                const char* xt = (const char*)x0 + (size_t)(t + 1) * 4;
                #pragma unroll
                for (int k = 0; k < 4; ++k)
                    a1s[xl[k]] = (__bf16)*(const float*)(xt + xi[k]);
            }
            #pragma unroll
            for (int d = 0; d < 8; ++d) {
                int m = d + half * 8;
                float v = acc[d] + bj3; v = v > 0.f ? v : 0.f;
                int ih = m * HS_STRIDE + j0 + l15;
                float h = 0.98f * hs[ih] + 0.01f * v;   // h + eta*(fn - 2h)
                hs[ih] = h;
                a1s[m * A1_STRIDE + j0 + l15] = (__bf16)(2.0f * h);
            }
        }
        __syncthreads();
    }

    // ---- epilogue: h1 out, then y1 = h1 @ Wout^T + b_out ----
    for (int i = tid; i < 16 * 256; i += 512) {
        int r = i >> 8, c = i & 255;
        out[65536 + (size_t)(b0 + r) * 256 + c] = hs[r * HS_STRIDE + c];
    }
    if (wave < 8) {
        int j0y = wave * 16;   // 8 waves cover O = 128 cols
        const __bf16* wO = wsb + WS_WOUT + (size_t)(j0y + l15) * 256 + kB; // 0.5*Wout
        v8f acc = {};
        #pragma unroll
        for (int kt = 0; kt < 8; ++kt) {
            Frag a, b;
            loadA(a, aH + kt * 32);    // 2h shadow; Wout pre-scaled by 0.5
            loadB(b, wO + kt * 32);
            WMMA(acc, a, b);
        }
        float bj = b_out[j0y + l15];
        #pragma unroll
        for (int d = 0; d < 8; ++d) {
            int m = d + half * 8;
            out[(size_t)(b0 + m) * 128 + j0y + l15] = acc[d] + bj;
        }
    }
}

extern "C" void kernel_launch(void* const* d_in, const int* in_sizes, int n_in,
                              void* d_out, int out_size, void* d_ws, size_t ws_size,
                              hipStream_t stream) {
    const float* x0    = (const float*)d_in[0];
    const float* h0    = (const float*)d_in[1];
    const float* W_in  = (const float*)d_in[2];
    const float* b_in  = (const float*)d_in[3];
    const float* W_hid = (const float*)d_in[4];
    const float* b_hid = (const float*)d_in[5];
    const float* W2    = (const float*)d_in[6];
    const float* b2    = (const float*)d_in[7];
    const float* W3    = (const float*)d_in[8];
    const float* b3    = (const float*)d_in[9];
    const float* W_out = (const float*)d_in[10];
    const float* b_out = (const float*)d_in[11];
    __bf16* wsb = (__bf16*)d_ws;   // needs 512 KiB
    float* out = (float*)d_out;

    prep_weights<<<WS_TOTAL / 256, 256, 0, stream>>>(W_hid, W2, W3, W_in, W_out, wsb);
    ernn_scan<<<32, 512, 0, stream>>>(x0, h0, b_in, b_hid, b2, b3, b_out, wsb, out);
}
